// MultiHeadSelfAttention_84877143703963
// MI455X (gfx1250) — compile-verified, matrix operand order checked by reading, not run on hardware
//
#include <hip/hip_runtime.h>
#include <hip/hip_bf16.h>
#include <math.h>

typedef __bf16 bf16_t;
typedef __bf16 v16bf __attribute__((ext_vector_type(16)));
typedef __bf16 bf8   __attribute__((ext_vector_type(8)));
typedef float  v8f   __attribute__((ext_vector_type(8)));

#define D_MODEL 1024
#define S_LEN   2048
#define BATCH   2
#define HEADS   16
#define DH      64
#define MROWS   (BATCH * S_LEN)   // 4096

union FragBF { v16bf v; bf8 h[2]; };

__device__ __forceinline__ bf8 load8(const bf16_t* p) { return *(const bf8*)p; }

// CDNA5 async global->LDS copy (per-lane 16B), tracked by ASYNCcnt.
__device__ __forceinline__ void async_b128(bf16_t* l, const bf16_t* g) {
    asm volatile("global_load_async_to_lds_b128 %0, %1, off"
                 :: "v"((uint32_t)(uintptr_t)l), "v"((uint64_t)(uintptr_t)g)
                 : "memory");
}
__device__ __forceinline__ void wait_async0() {
    asm volatile("s_wait_asynccnt 0" ::: "memory");
}

// ---------------- conversion kernels ----------------

__global__ void k_cvt_x(const float* __restrict__ x, bf16_t* __restrict__ xb, int n) {
    int i = blockIdx.x * blockDim.x + threadIdx.x;
    if (i < n) xb[i] = (bf16_t)x[i];
}

// Wt[n*1024 + k] = W[k*1024 + n]  (store W^T in bf16)
__global__ void k_cvt_wT(const float* __restrict__ W, bf16_t* __restrict__ Wt) {
    int i = blockIdx.x * blockDim.x + threadIdx.x;
    int n = i >> 10, k = i & 1023;
    Wt[i] = (bf16_t)W[k * D_MODEL + n];
}

// ---------------- bf16 WMMA GEMM: out = A(MxK) * B(KxN) + bias ----------------
// A row-major MxK bf16; Bt = B^T row-major NxK bf16.
// mode 0: bf16 out, head-major [B,H,S,dh]      (for Q, K)
// mode 1: bf16 out, transposed [B,H,dh,S]      (for V)
// mode 2: f32 out, row-major [M,N]             (final projection)
__global__ __launch_bounds__(256) void k_gemm(const bf16_t* __restrict__ A,
                                              const bf16_t* __restrict__ Bt,
                                              const float* __restrict__ bias,
                                              void* __restrict__ out, int mode) {
    __shared__ __align__(16) bf16_t As[2][128 * 32];
    __shared__ __align__(16) bf16_t Bs[2][128 * 32];
    const int t = threadIdx.x;
    const int wave = t >> 5, lane = t & 31;
    const int ln = lane & 15, hf = lane >> 4;
    const int wm = wave >> 2, wn = wave & 3;   // 2 x 4 wave grid -> wave tile 64x32
    const int m0 = blockIdx.x * 128;
    const int n0 = blockIdx.y * 128;

    auto stage = [&](int buf, int kk) {
        #pragma unroll
        for (int i = 0; i < 2; i++) {
            int cc = t + i * 256;
            int r = cc >> 2, ch = cc & 3;
            async_b128(&As[buf][r * 32 + ch * 8],
                       &A [(size_t)(m0 + r) * D_MODEL + kk + ch * 8]);
            async_b128(&Bs[buf][r * 32 + ch * 8],
                       &Bt[(size_t)(n0 + r) * D_MODEL + kk + ch * 8]);
        }
    };

    v8f acc[4][2];
    #pragma unroll
    for (int i = 0; i < 4; i++)
        #pragma unroll
        for (int j = 0; j < 2; j++) acc[i][j] = v8f{};

    stage(0, 0);
    wait_async0();
    __syncthreads();

    int buf = 0;
    for (int kk = 0; kk < D_MODEL; kk += 32, buf ^= 1) {
        if (kk + 32 < D_MODEL) stage(buf ^ 1, kk + 32);   // overlap next-tile fill

        FragBF a[4], b[2];
        #pragma unroll
        for (int i = 0; i < 4; i++) {
            int r = wm * 64 + i * 16 + ln;
            a[i].h[0] = *(const bf8*)&As[buf][r * 32 + 8 * hf];
            a[i].h[1] = *(const bf8*)&As[buf][r * 32 + 16 + 8 * hf];
        }
        #pragma unroll
        for (int j = 0; j < 2; j++) {
            int r = wn * 32 + j * 16 + ln;
            b[j].h[0] = *(const bf8*)&Bs[buf][r * 32 + 8 * hf];
            b[j].h[1] = *(const bf8*)&Bs[buf][r * 32 + 16 + 8 * hf];
        }
        #pragma unroll
        for (int i = 0; i < 4; i++)
            #pragma unroll
            for (int j = 0; j < 2; j++)
                acc[i][j] = __builtin_amdgcn_wmma_f32_16x16x32_bf16(
                    false, a[i].v, false, b[j].v, (short)0, acc[i][j], false, false);

        wait_async0();
        __syncthreads();
    }

    #pragma unroll
    for (int i = 0; i < 4; i++)
        #pragma unroll
        for (int j = 0; j < 2; j++)
            #pragma unroll
            for (int e = 0; e < 8; e++) {
                int gm = m0 + wm * 64 + i * 16 + e + 8 * hf;
                int gn = n0 + wn * 32 + j * 16 + ln;
                float v = acc[i][j][e] + bias[gn];
                if (mode == 2) {
                    ((float*)out)[(size_t)gm * D_MODEL + gn] = v;
                } else {
                    int bb = gm >> 11, s = gm & 2047;
                    int hh = gn >> 6,  d = gn & 63;
                    bf16_t* ob = (bf16_t*)out;
                    if (mode == 0)
                        ob[(((size_t)(bb * HEADS + hh) * S_LEN) + s) * DH + d] = (bf16_t)v;
                    else
                        ob[((size_t)(bb * HEADS + hh) * DH + d) * S_LEN + s] = (bf16_t)v;
                }
            }
}

// ---------------- flash attention (bf16 WMMA, f32 softmax) ----------------
// Qh, Kh: [B*H][S][dh] bf16.  Vt: [B*H][dh][S] bf16.  O: [B][S][D] bf16.
__global__ __launch_bounds__(256) void k_attn(const bf16_t* __restrict__ Qh,
                                              const bf16_t* __restrict__ Kh,
                                              const bf16_t* __restrict__ Vt,
                                              bf16_t* __restrict__ O) {
    __shared__ __align__(16) bf16_t Ks[2][32 * 64];    // K tile: 32 keys x 64 d
    __shared__ __align__(16) bf16_t Vs[2][64 * 32];    // V^T tile: 64 d x 32 keys
    __shared__ __align__(16) bf16_t Ps[8 * 16 * 32];   // per-wave P tile (16 q x 32 k)
    const int t = threadIdx.x;
    const int wave = t >> 5, lane = t & 31;
    const int ln = lane & 15, hf = lane >> 4;
    const int bh = blockIdx.x;                 // 0..31
    const int qb = blockIdx.y;                 // 0..15
    const int qBase = qb * 128 + wave * 16;

    const bf16_t* Qg = Qh + (size_t)bh * S_LEN * DH;
    const bf16_t* Kg = Kh + (size_t)bh * S_LEN * DH;
    const bf16_t* Vg = Vt + (size_t)bh * DH * S_LEN;

    auto stageKV = [&](int buf, int k0) {
        { int r = t >> 3, ch = t & 7;
          async_b128(&Ks[buf][r * 64 + ch * 8], &Kg[(size_t)(k0 + r) * DH + ch * 8]); }
        { int r = t >> 2, ch = t & 3;
          async_b128(&Vs[buf][r * 32 + ch * 8], &Vg[(size_t)r * S_LEN + k0 + ch * 8]); }
    };

    FragBF qf[2];
    {
        const bf16_t* qr = Qg + (size_t)(qBase + ln) * DH;
        qf[0].h[0] = load8(qr + 8 * hf);
        qf[0].h[1] = load8(qr + 16 + 8 * hf);
        qf[1].h[0] = load8(qr + 32 + 8 * hf);
        qf[1].h[1] = load8(qr + 48 + 8 * hf);
    }

    v8f o0 = v8f{}, o1 = v8f{}, o2 = v8f{}, o3 = v8f{};
    float mrow[8], lrow[8];
    #pragma unroll
    for (int j = 0; j < 8; j++) { mrow[j] = -INFINITY; lrow[j] = 0.0f; }

    const float invScale = 1.0f / (sqrtf((float)D_MODEL) + 1e-9f);
    bf16_t* pw = &Ps[wave * (16 * 32)];

    stageKV(0, 0);
    wait_async0();
    __syncthreads();

    int buf = 0;
    for (int k0 = 0; k0 < S_LEN; k0 += 32, buf ^= 1) {
        if (k0 + 32 < S_LEN) stageKV(buf ^ 1, k0 + 32);   // overlap next-tile fill

        // S = Q * K^T for two 16-key subtiles (contraction over dh=64 = 2 WMMAs each)
        v8f sc[2];
        #pragma unroll
        for (int kt = 0; kt < 2; kt++) {
            FragBF kf0, kf1;
            int r = kt * 16 + ln;
            kf0.h[0] = *(const bf8*)&Ks[buf][r * 64 + 8 * hf];
            kf0.h[1] = *(const bf8*)&Ks[buf][r * 64 + 16 + 8 * hf];
            kf1.h[0] = *(const bf8*)&Ks[buf][r * 64 + 32 + 8 * hf];
            kf1.h[1] = *(const bf8*)&Ks[buf][r * 64 + 48 + 8 * hf];
            v8f c = v8f{};
            c = __builtin_amdgcn_wmma_f32_16x16x32_bf16(false, qf[0].v, false, kf0.v, (short)0, c, false, false);
            c = __builtin_amdgcn_wmma_f32_16x16x32_bf16(false, qf[1].v, false, kf1.v, (short)0, c, false, false);
            sc[kt] = c;
        }

        // online softmax: rows live across 16 lanes of each half-wave
        float p0[8], p1[8];
        #pragma unroll
        for (int j = 0; j < 8; j++) {
            float s0 = sc[0][j] * invScale, s1 = sc[1][j] * invScale;
            float rm = fmaxf(s0, s1);
            rm = fmaxf(rm, __shfl_xor(rm, 1, 32));
            rm = fmaxf(rm, __shfl_xor(rm, 2, 32));
            rm = fmaxf(rm, __shfl_xor(rm, 4, 32));
            rm = fmaxf(rm, __shfl_xor(rm, 8, 32));
            float mnew  = fmaxf(mrow[j], rm);
            float alpha = __expf(mrow[j] - mnew);
            float e0 = __expf(s0 - mnew);
            float e1 = __expf(s1 - mnew);
            float rs = e0 + e1;
            rs += __shfl_xor(rs, 1, 32);
            rs += __shfl_xor(rs, 2, 32);
            rs += __shfl_xor(rs, 4, 32);
            rs += __shfl_xor(rs, 8, 32);
            lrow[j] = lrow[j] * alpha + rs;
            mrow[j] = mnew;
            p0[j] = e0; p1[j] = e1;
            o0[j] = o0[j] * alpha; o1[j] = o1[j] * alpha;
            o2[j] = o2[j] * alpha; o3[j] = o3[j] * alpha;
        }

        // P (C/D layout) -> LDS row-major 16x32 bf16, then reload as A fragment
        #pragma unroll
        for (int j = 0; j < 8; j++) {
            int row = j + 8 * hf;
            pw[row * 32 + ln]      = (bf16_t)p0[j];
            pw[row * 32 + 16 + ln] = (bf16_t)p1[j];
        }
        asm volatile("s_wait_dscnt 0" ::: "memory");
        FragBF pf;
        pf.h[0] = *(const bf8*)&pw[ln * 32 + 8 * hf];
        pf.h[1] = *(const bf8*)&pw[ln * 32 + 16 + 8 * hf];

        // O += P * V  (4 d-chunks of 16)
        #pragma unroll
        for (int n = 0; n < 4; n++) {
            FragBF vf;
            int r = n * 16 + ln;
            vf.h[0] = *(const bf8*)&Vs[buf][r * 32 + 8 * hf];
            vf.h[1] = *(const bf8*)&Vs[buf][r * 32 + 16 + 8 * hf];
            v8f* op = (n == 0) ? &o0 : (n == 1) ? &o1 : (n == 2) ? &o2 : &o3;
            *op = __builtin_amdgcn_wmma_f32_16x16x32_bf16(false, pf.v, false, vf.v, (short)0, *op, false, false);
        }

        wait_async0();
        __syncthreads();
    }

    // normalize and write O[b][s][head*64 + d] in bf16
    const int b = bh >> 4, head = bh & 15;
    #pragma unroll
    for (int j = 0; j < 8; j++) {
        int s = qBase + j + 8 * hf;
        float rinv = (lrow[j] > 0.0f) ? 1.0f / lrow[j] : 0.0f;
        size_t base = ((size_t)(b * S_LEN + s)) * D_MODEL + head * DH;
        O[base + 0 * 16 + ln] = (bf16_t)(o0[j] * rinv);
        O[base + 1 * 16 + ln] = (bf16_t)(o1[j] * rinv);
        O[base + 2 * 16 + ln] = (bf16_t)(o2[j] * rinv);
        O[base + 3 * 16 + ln] = (bf16_t)(o3[j] * rinv);
    }
}

// ---------------- launch ----------------

extern "C" void kernel_launch(void* const* d_in, const int* in_sizes, int n_in,
                              void* d_out, int out_size, void* d_ws, size_t ws_size,
                              hipStream_t stream) {
    const float* x  = (const float*)d_in[0];
    const float* WQ = (const float*)d_in[1];
    const float* bQ = (const float*)d_in[2];
    const float* WK = (const float*)d_in[3];
    const float* bK = (const float*)d_in[4];
    const float* WV = (const float*)d_in[5];
    const float* bV = (const float*)d_in[6];
    const float* WO = (const float*)d_in[7];
    const float* bO = (const float*)d_in[8];

    char* ws = (char*)d_ws;
    size_t off = 0;
    auto alloc = [&](size_t bytes) -> void* {
        void* p = ws + off;
        off += (bytes + 255) & ~(size_t)255;
        return p;
    };
    bf16_t* xb  = (bf16_t*)alloc((size_t)MROWS * D_MODEL * 2);
    bf16_t* WQt = (bf16_t*)alloc((size_t)D_MODEL * D_MODEL * 2);
    bf16_t* WKt = (bf16_t*)alloc((size_t)D_MODEL * D_MODEL * 2);
    bf16_t* WVt = (bf16_t*)alloc((size_t)D_MODEL * D_MODEL * 2);
    bf16_t* WOt = (bf16_t*)alloc((size_t)D_MODEL * D_MODEL * 2);
    bf16_t* Qh  = (bf16_t*)alloc((size_t)MROWS * D_MODEL * 2);
    bf16_t* Kh  = (bf16_t*)alloc((size_t)MROWS * D_MODEL * 2);
    bf16_t* Vh  = (bf16_t*)alloc((size_t)MROWS * D_MODEL * 2);
    bf16_t* Ob  = (bf16_t*)alloc((size_t)MROWS * D_MODEL * 2);

    const int NX = MROWS * D_MODEL;           // 4M
    k_cvt_x<<<(NX + 255) / 256, 256, 0, stream>>>(x, xb, NX);
    const int NW = D_MODEL * D_MODEL;         // 1M
    k_cvt_wT<<<NW / 256, 256, 0, stream>>>(WQ, WQt);
    k_cvt_wT<<<NW / 256, 256, 0, stream>>>(WK, WKt);
    k_cvt_wT<<<NW / 256, 256, 0, stream>>>(WV, WVt);
    k_cvt_wT<<<NW / 256, 256, 0, stream>>>(WO, WOt);

    dim3 ggrid(MROWS / 128, D_MODEL / 128);   // 32 x 8
    k_gemm<<<ggrid, 256, 0, stream>>>(xb, WQt, bQ, (void*)Qh, 0);
    k_gemm<<<ggrid, 256, 0, stream>>>(xb, WKt, bK, (void*)Kh, 0);
    k_gemm<<<ggrid, 256, 0, stream>>>(xb, WVt, bV, (void*)Vh, 1);

    dim3 agrid(BATCH * HEADS, S_LEN / 128);   // 32 x 16
    k_attn<<<agrid, 256, 0, stream>>>(Qh, Kh, Vh, Ob);

    k_gemm<<<ggrid, 256, 0, stream>>>(Ob, WOt, bO, d_out, 2);
}
